// DecoyPairStackBlock_56813827392304
// MI455X (gfx1250) — matrix operand order repaired and checked by reading, not compiled, not run on hardware
//
#include <hip/hip_runtime.h>
#include <hip/hip_fp16.h>

typedef _Float16 f16;
typedef __attribute__((ext_vector_type(16))) _Float16 v16h;
typedef __attribute__((ext_vector_type(8)))  float    v8f;

#define NN 256
#define CC 64
#define RTOT 262144  // T*N*N = 4*256*256

// ---------------------------------------------------------------------------
// WMMA helpers (gfx1250, wave32). A-frag layout used here (16-bit A 16x32):
//   lane = m (0..15) in each half-wave, element e -> K = 16*kh + e
// B-frag layout assumed sequential per half-wave: element e -> K = 16*kh + e
// C/D layout: acc[q] -> row = q + 8*kh, col = lane&15
// ---------------------------------------------------------------------------
__device__ __forceinline__ v8f wmma32(v16h a, v16h b, v8f c) {
  return __builtin_amdgcn_wmma_f32_16x16x32_f16(false, a, false, b, (short)0, c,
                                                false, false);
}
__device__ __forceinline__ int ka_idx(int e, int kh) {
  return kh * 16 + e;
}
__device__ __forceinline__ v16h fragA(const f16* p, int kh) {
  v16h v;
#pragma unroll
  for (int e = 0; e < 16; ++e) v[e] = p[ka_idx(e, kh)];
  return v;
}
__device__ __forceinline__ v16h fragB(const f16* p, int kh) {
  v16h v;
#pragma unroll
  for (int e = 0; e < 16; ++e) v[e] = p[kh * 16 + e];
  return v;
}
__device__ __forceinline__ float sigm(float x) { return 1.f / (1.f + expf(-x)); }

// ---------------------------------------------------------------------------
// CDNA5 async global->LDS copy (ASYNCcnt-tracked, bypasses VGPRs).
// LDS offset = low 32 bits of the flat shared-memory address (ISA 10.2: the
// LDS aperture maps addr[31:0] directly to the LDS byte address).
// ---------------------------------------------------------------------------
__device__ __forceinline__ void async_g2l_b128(void* lds_dst,
                                               const void* gsrc) {
  unsigned loff = (unsigned)(unsigned long long)lds_dst;
  asm volatile("global_load_async_to_lds_b128 %0, %1, off" ::"v"(loff),
               "v"(gsrc)
               : "memory");
}
__device__ __forceinline__ void wait_async0() {
  asm volatile("s_wait_asynccnt 0" ::: "memory");
}

// ---------------------------------------------------------------------------
// LayerNorm over C=64 (+ optional triangle-bias x@wb^T), optional transposed
// read of z (for the "ending node" attention). 16 lanes per row, 16 rows/block.
// ---------------------------------------------------------------------------
__global__ __launch_bounds__(256) void ln_kernel(
    const float* __restrict__ in, const float* __restrict__ gam,
    const float* __restrict__ bet, const float* __restrict__ wb,
    f16* __restrict__ xout, float* __restrict__ tb, int transposed) {
  int row = blockIdx.x * 16 + (threadIdx.x >> 4);
  int l = threadIdx.x & 15;
  int c0 = l * 4;
  size_t src;
  if (transposed) {
    int t = row >> 16, i1 = (row >> 8) & 255, i2 = row & 255;
    src = ((size_t)((t * NN + i2) * NN + i1)) * CC + c0;
  } else {
    src = (size_t)row * CC + c0;
  }
  float4 x = *(const float4*)(in + src);
  float s = x.x + x.y + x.z + x.w;
  float ss = x.x * x.x + x.y * x.y + x.z * x.z + x.w * x.w;
#pragma unroll
  for (int m = 1; m < 16; m <<= 1) {
    s += __shfl_xor(s, m, 16);
    ss += __shfl_xor(ss, m, 16);
  }
  float mean = s * (1.f / 64.f);
  float var = ss * (1.f / 64.f) - mean * mean;
  float rs = rsqrtf(var + 1e-5f);
  float xv[4] = {x.x, x.y, x.z, x.w};
  float y[4];
#pragma unroll
  for (int j = 0; j < 4; ++j) {
    y[j] = (xv[j] - mean) * rs * gam[c0 + j] + bet[c0 + j];
    xout[(size_t)row * CC + c0 + j] = (f16)y[j];
  }
  if (wb) {
    float th0 = 0, th1 = 0, th2 = 0, th3 = 0;
#pragma unroll
    for (int j = 0; j < 4; ++j) {
      th0 += y[j] * wb[0 * CC + c0 + j];
      th1 += y[j] * wb[1 * CC + c0 + j];
      th2 += y[j] * wb[2 * CC + c0 + j];
      th3 += y[j] * wb[3 * CC + c0 + j];
    }
#pragma unroll
    for (int m = 1; m < 16; m <<= 1) {
      th0 += __shfl_xor(th0, m, 16);
      th1 += __shfl_xor(th1, m, 16);
      th2 += __shfl_xor(th2, m, 16);
      th3 += __shfl_xor(th3, m, 16);
    }
    if (l < 4) {
      float v = (l == 0) ? th0 : (l == 1) ? th1 : (l == 2) ? th2 : th3;
      tb[(size_t)row * 4 + l] = v;
    }
  }
}

// ---------------------------------------------------------------------------
// Generic WMMA GEMM: Y[r,o] = sum_c X[r,c]*W[o,c], K in {64,128}, OUT in {64,128}
// 8 waves/block; wave -> one 16x16 (row x out) tile; weights f32->f16 in LDS,
// X tile staged via async global->LDS (overlaps with weight conversion).
// Epilogues fuse scale / sigmoid / relu / residual / gate / mask.
// ---------------------------------------------------------------------------
template <int EPI>
__global__ __launch_bounds__(256) void gemm_kernel(
    const f16* __restrict__ X, const float* __restrict__ W,
    const float* __restrict__ bias, int K, int OUT, float scale,
    f16* __restrict__ out16, float* __restrict__ zio,
    const f16* __restrict__ gate16, const float* __restrict__ maskp,
    int transposed) {
  extern __shared__ char smem[];
  f16* Wl = (f16*)smem;
  const int OT = OUT >> 4;
  const int RPB = 16 * (8 / OT);
  f16* Xl = Wl + OUT * K;
  size_t rowbase = (size_t)blockIdx.x * RPB;
  int nv = RPB * K / 8;
  const f16* xs = X + rowbase * K;
  for (int idx = threadIdx.x; idx < nv; idx += 256)
    async_g2l_b128(Xl + idx * 8, xs + idx * 8);
  for (int idx = threadIdx.x; idx < OUT * K; idx += 256) Wl[idx] = (f16)W[idx];
  wait_async0();
  __syncthreads();
  int w = threadIdx.x >> 5, lane = threadIdx.x & 31, r = lane & 15,
      kh = lane >> 4;
  int rt = w / OT, ot = w % OT;
  v8f acc = {};
  for (int kb = 0; kb < K; kb += 32) {
    v16h a = fragA(Xl + (rt * 16 + r) * K + kb, kh);
    v16h b = fragB(Wl + (ot * 16 + r) * K + kb, kh);
    acc = wmma32(a, b, acc);
  }
  int oc = ot * 16 + r;
  float bv = bias ? bias[oc] : 0.f;
#pragma unroll
  for (int q = 0; q < 8; ++q) {
    int rr = (int)rowbase + rt * 16 + q + 8 * kh;
    float v = acc[q];
    if (EPI == 0) {  // plain store f16 (scaled)
      out16[(size_t)rr * OUT + oc] = (f16)(v * scale);
    } else if (EPI == 1) {  // sigmoid gate store f16
      out16[(size_t)rr * OUT + oc] = (f16)sigm(v + bv);
    } else if (EPI == 2) {  // relu store f16
      out16[(size_t)rr * OUT + oc] = (f16)fmaxf(v + bv, 0.f);
    } else if (EPI == 3) {  // residual add into z (optionally transposed)
      size_t di;
      if (transposed) {
        int t = rr >> 16, i1 = (rr >> 8) & 255, i2 = rr & 255;
        di = ((size_t)((t * NN + i2) * NN + i1)) * CC + oc;
      } else {
        di = (size_t)rr * CC + oc;
      }
      zio[di] += v + bv;
    } else if (EPI == 4) {  // gated residual: z += (y+b)*g
      size_t di = (size_t)rr * CC + oc;
      zio[di] += (v + bv) * (float)gate16[di];
    } else if (EPI == 5) {  // masked residual: z += (y+b)*mask
      zio[(size_t)rr * CC + oc] += (v + bv) * maskp[rr];
    }
  }
}

// ---------------------------------------------------------------------------
// Fused dual-GEMM for triangle-mul projections:
//   out = mask * sigmoid(x@Wg^T + bg) * (x@Wp^T + bp)
// stored channel-major [t, c, s1, s2] (spatially transposed for "incoming").
// ---------------------------------------------------------------------------
__global__ __launch_bounds__(256) void gatemul_kernel(
    const f16* __restrict__ X, const float* __restrict__ Wg,
    const float* __restrict__ bg, const float* __restrict__ Wp,
    const float* __restrict__ bp, const float* __restrict__ maskp,
    f16* __restrict__ outab, int transposed) {
  extern __shared__ char smem[];
  f16* Wgl = (f16*)smem;
  f16* Wpl = Wgl + 4096;
  f16* Xl = Wpl + 4096;
  size_t rowbase = (size_t)blockIdx.x * 32;
  const f16* xs = X + rowbase * 64;
  for (int idx = threadIdx.x; idx < 256; idx += 256)
    async_g2l_b128(Xl + idx * 8, xs + idx * 8);
  for (int idx = threadIdx.x; idx < 4096; idx += 256) {
    Wgl[idx] = (f16)Wg[idx];
    Wpl[idx] = (f16)Wp[idx];
  }
  wait_async0();
  __syncthreads();
  int w = threadIdx.x >> 5, lane = threadIdx.x & 31, r = lane & 15,
      kh = lane >> 4;
  int rt = w >> 2, ot = w & 3;
  v8f ag = {}, ap = {};
  for (int kb = 0; kb < 64; kb += 32) {
    v16h a = fragA(Xl + (rt * 16 + r) * 64 + kb, kh);
    ag = wmma32(a, fragB(Wgl + (ot * 16 + r) * 64 + kb, kh), ag);
    ap = wmma32(a, fragB(Wpl + (ot * 16 + r) * 64 + kb, kh), ap);
  }
  int c = ot * 16 + r;
  float bgv = bg[c], bpv = bp[c];
#pragma unroll
  for (int q = 0; q < 8; ++q) {
    int rr = (int)rowbase + rt * 16 + q + 8 * kh;
    int t = rr >> 16, i1 = (rr >> 8) & 255, i2 = rr & 255;
    float val = maskp[rr] * sigm(ag[q] + bgv) * (ap[q] + bpv);
    size_t di = transposed ? ((size_t)((t * 64 + c) * NN + i2)) * NN + i1
                           : ((size_t)((t * 64 + c) * NN + i1)) * NN + i2;
    outab[di] = (f16)val;
  }
}

// ---------------------------------------------------------------------------
// Triangle einsum: per (t,c) slab, T[i,j] = sum_k A[i,k]*B[j,k]  (256^3 GEMM)
// A,B channel-major f16 [slab][256][256] staged to LDS via async copies.
// ---------------------------------------------------------------------------
__global__ __launch_bounds__(256) void einsum_kernel(const f16* __restrict__ A,
                                                     const f16* __restrict__ B,
                                                     float* __restrict__ Tt) {
  extern __shared__ char smem[];
  f16* Al = (f16*)smem;      // 64 x 256
  f16* Bl = Al + 64 * 256;   // 32 x 256
  int jb = blockIdx.x, ib = blockIdx.y, s = blockIdx.z;
  const f16* As = A + (size_t)s * 65536 + ib * 64 * 256;
  const f16* Bs = B + (size_t)s * 65536 + jb * 32 * 256;
  for (int idx = threadIdx.x; idx < 2048; idx += 256)
    async_g2l_b128(Al + idx * 8, As + idx * 8);
  for (int idx = threadIdx.x; idx < 1024; idx += 256)
    async_g2l_b128(Bl + idx * 8, Bs + idx * 8);
  wait_async0();
  __syncthreads();
  int w = threadIdx.x >> 5, lane = threadIdx.x & 31, r = lane & 15,
      kh = lane >> 4;
  int it = w & 3, jt = w >> 2;
  v8f acc = {};
  for (int kb = 0; kb < 256; kb += 32) {
    v16h a = fragA(Al + (it * 16 + r) * 256 + kb, kh);
    v16h b = fragB(Bl + (jt * 16 + r) * 256 + kb, kh);
    acc = wmma32(a, b, acc);
  }
  int t = s >> 6, c = s & 63;
  int j = jb * 32 + jt * 16 + r;
#pragma unroll
  for (int q = 0; q < 8; ++q) {
    int i = ib * 64 + it * 16 + q + 8 * kh;
    Tt[((size_t)((t * NN + i) * NN + j)) * CC + c] = acc[q];
  }
}

// ---------------------------------------------------------------------------
// Triangle attention core. One block per (t, i, head, 64-row q-block).
// 4 waves x 16 q-rows. Scores via WMMA (d=16 zero-padded into K=32 halves),
// bias+mask fused, split-half softmax (all 32 lanes), P@V via WMMA,
// sigmoid-gate fused at store. K/V staged via async global->LDS.
// ---------------------------------------------------------------------------
__global__ __launch_bounds__(128) void attn_kernel(
    const f16* __restrict__ qh, const f16* __restrict__ khp,
    const f16* __restrict__ vh, const float* __restrict__ tb,
    const f16* __restrict__ gh, const float* __restrict__ maskp,
    f16* __restrict__ obuf, int transposed) {
  extern __shared__ char smem[];
  f16* Kl = (f16*)smem;                 // 256 x 16
  f16* Vl = Kl + 256 * 16;              // 256 x 16
  float* Sl = (float*)(Vl + 256 * 16);  // 4 waves x 16 x 256 f32
  f16* Pl = (f16*)(Sl + 4 * 16 * 256);  // 4 waves x 16 x 256 f16
  float* inv = (float*)(Pl + 4 * 16 * 256);  // 4 x 16

  int b = blockIdx.x;
  int qb = b & 3, h = (b >> 2) & 3, i = (b >> 4) & 255, t = b >> 12;
  size_t base = (size_t)(t * NN + i) * NN;

  for (int kk = threadIdx.x; kk < 256; kk += 128) {
    size_t src = (base + kk) * CC + h * 16;
    async_g2l_b128(Kl + kk * 16, khp + src);
    async_g2l_b128(Kl + kk * 16 + 8, khp + src + 8);
    async_g2l_b128(Vl + kk * 16, vh + src);
    async_g2l_b128(Vl + kk * 16 + 8, vh + src + 8);
  }
  wait_async0();
  __syncthreads();

  int w = threadIdx.x >> 5, lane = threadIdx.x & 31, r = lane & 15,
      kh = lane >> 4;
  int q0 = qb * 64 + w * 16;
  float* Sw = Sl + w * 16 * 256;
  f16* Pw = Pl + w * 16 * 256;
  float* invw = inv + w * 16;

  v16h qa;
  {
    const f16* qp = qh + (base + q0 + r) * CC + h * 16;
#pragma unroll
    for (int e = 0; e < 16; ++e) {
      int kk = ka_idx(e, kh);
      qa[e] = (kk < 16) ? qp[kk] : (f16)0.0f;  // q pre-scaled by 1/sqrt(16)
    }
  }
  for (int kt = 0; kt < 16; ++kt) {
    v16h bb;
#pragma unroll
    for (int e = 0; e < 16; ++e) {
      int kk = kh * 16 + e;
      bb[e] = (kk < 16) ? Kl[(kt * 16 + r) * 16 + kk] : (f16)0.0f;
    }
    v8f c = {};
    c = wmma32(qa, bb, c);
    int kcol = kt * 16 + r;
    float mv = transposed ? maskp[((size_t)t * NN + kcol) * NN + i]
                          : maskp[((size_t)t * NN + i) * NN + kcol];
    float mbias = 1e9f * (mv - 1.f);
#pragma unroll
    for (int q = 0; q < 8; ++q) {
      int row = q + 8 * kh;
      float bias =
          mbias + tb[((size_t)(t * NN + (q0 + row)) * NN + kcol) * 4 + h];
      Sw[row * 256 + kcol] = c[q] + bias;
    }
  }
  __syncthreads();
  {
    // split-half softmax: lane r handles cols 0..127 of row r, lane r+16
    // handles cols 128..255; combine with a cross-half shuffle.
    int row = r;
    int cb = kh * 128;
    float mx = -1e30f;
    for (int k2 = 0; k2 < 128; ++k2)
      mx = fmaxf(mx, Sw[row * 256 + cb + k2]);
    mx = fmaxf(mx, __shfl_xor(mx, 16, 32));
    float s = 0.f;
    for (int k2 = 0; k2 < 128; ++k2) {
      float e = expf(Sw[row * 256 + cb + k2] - mx);
      s += e;
      Pw[row * 256 + cb + k2] = (f16)e;
    }
    s += __shfl_xor(s, 16, 32);
    if (kh == 0) invw[row] = 1.f / s;
  }
  __syncthreads();
  v8f acc = {};
  for (int ck = 0; ck < 8; ++ck) {
    v16h pa, vb;
#pragma unroll
    for (int e = 0; e < 16; ++e) {
      pa[e] = Pw[r * 256 + ck * 32 + ka_idx(e, kh)];
      vb[e] = Vl[(ck * 32 + kh * 16 + e) * 16 + r];
    }
    acc = wmma32(pa, vb, acc);
  }
#pragma unroll
  for (int q = 0; q < 8; ++q) {
    int row = q + 8 * kh;
    size_t rr = base + (q0 + row);
    float val = acc[q] * invw[row];
    float g = (float)gh[rr * CC + h * 16 + r];
    obuf[rr * CC + h * 16 + r] = (f16)(val * g);
  }
}

// ---------------------------------------------------------------------------
// Host orchestration
// ---------------------------------------------------------------------------
extern "C" void kernel_launch(void* const* d_in, const int* in_sizes, int n_in,
                              void* d_out, int out_size, void* d_ws,
                              size_t ws_size, hipStream_t stream) {
  const float* zin = (const float*)d_in[0];
  const float* maskp = (const float*)d_in[1];
  auto P = [&](int i) { return (const float*)d_in[i]; };
  float* zout = (float*)d_out;
  char* ws = (char*)d_ws;
  const size_t BUF = 33554432ull;  // 16.7M f16 elems
  f16* b0 = (f16*)(ws);            // xh / th
  f16* b1 = (f16*)(ws + BUF);      // qh / abuf
  f16* b2 = (f16*)(ws + 2 * BUF);  // kh / bbuf
  f16* b3 = (f16*)(ws + 3 * BUF);  // vh / gbuf
  f16* b4 = (f16*)(ws + 4 * BUF);  // gh ; start of tbuf/hbuf f32 region
  f16* b5 = (f16*)(ws + 5 * BUF);  // obuf
  float* tbuf = (float*)(ws + 4 * BUF);
  float* tbp = (float*)(ws + 6 * BUF);  // 4MB triangle-bias

  hipMemcpyAsync(zout, zin, (size_t)RTOT * CC * sizeof(float),
                 hipMemcpyDeviceToDevice, stream);

  auto gemm = [&](int epi, const f16* X, const float* W, const float* bias,
                  int K, int OUT, float scale, f16* o16, float* zio,
                  const f16* g16, const float* mp, int tr) {
    int RPB = 16 * (8 / (OUT / 16));
    int blocks = RTOT / RPB;
    size_t sm = (size_t)OUT * K * 2 + (size_t)RPB * K * 2;
    switch (epi) {
      case 0: gemm_kernel<0><<<blocks, 256, sm, stream>>>(X, W, bias, K, OUT, scale, o16, zio, g16, mp, tr); break;
      case 1: gemm_kernel<1><<<blocks, 256, sm, stream>>>(X, W, bias, K, OUT, scale, o16, zio, g16, mp, tr); break;
      case 2: gemm_kernel<2><<<blocks, 256, sm, stream>>>(X, W, bias, K, OUT, scale, o16, zio, g16, mp, tr); break;
      case 3: gemm_kernel<3><<<blocks, 256, sm, stream>>>(X, W, bias, K, OUT, scale, o16, zio, g16, mp, tr); break;
      case 4: gemm_kernel<4><<<blocks, 256, sm, stream>>>(X, W, bias, K, OUT, scale, o16, zio, g16, mp, tr); break;
      case 5: gemm_kernel<5><<<blocks, 256, sm, stream>>>(X, W, bias, K, OUT, scale, o16, zio, g16, mp, tr); break;
    }
  };

  auto tri_att = [&](int pb, int tr) {
    // pb+0 ln_g, +1 ln_b, +2 wq, +3 wk, +4 wv, +5 wb, +6 wg, +7 bg, +8 wo, +9 bo
    ln_kernel<<<RTOT / 16, 256, 0, stream>>>(zout, P(pb), P(pb + 1), P(pb + 5),
                                             b0, tbp, tr);
    gemm(0, b0, P(pb + 2), nullptr, 64, 64, 0.25f, b1, nullptr, nullptr, nullptr, 0);
    gemm(0, b0, P(pb + 3), nullptr, 64, 64, 1.0f, b2, nullptr, nullptr, nullptr, 0);
    gemm(0, b0, P(pb + 4), nullptr, 64, 64, 1.0f, b3, nullptr, nullptr, nullptr, 0);
    gemm(1, b0, P(pb + 6), P(pb + 7), 64, 64, 1.0f, b4, nullptr, nullptr, nullptr, 0);
    size_t asm_sz = 16384 + 65536 + 32768 + 256;
    attn_kernel<<<16384, 128, asm_sz, stream>>>(b1, b2, b3, tbp, b4, maskp, b5, tr);
    gemm(3, b5, P(pb + 8), P(pb + 9), 64, 64, 1.0f, nullptr, zout, nullptr, nullptr, tr);
  };

  auto tri_mul = [&](int pb, int ts) {
    // +0 ln_in_g, +1 ln_in_b, +2 ln_out_g, +3 ln_out_b, +4 w_ap, +5 b_ap,
    // +6 w_ag, +7 b_ag, +8 w_bp, +9 b_bp, +10 w_bg, +11 b_bg, +12 w_g, +13 b_g,
    // +14 w_z, +15 b_z
    ln_kernel<<<RTOT / 16, 256, 0, stream>>>(zout, P(pb), P(pb + 1), nullptr,
                                             b0, nullptr, 0);
    gatemul_kernel<<<RTOT / 32, 256, 20480, stream>>>(
        b0, P(pb + 6), P(pb + 7), P(pb + 4), P(pb + 5), maskp, b1, ts);
    gatemul_kernel<<<RTOT / 32, 256, 20480, stream>>>(
        b0, P(pb + 10), P(pb + 11), P(pb + 8), P(pb + 9), maskp, b2, ts);
    gemm(1, b0, P(pb + 12), P(pb + 13), 64, 64, 1.0f, b3, nullptr, nullptr, nullptr, 0);
    einsum_kernel<<<dim3(8, 4, 256), 256, 49152, stream>>>(b1, b2, tbuf);
    ln_kernel<<<RTOT / 16, 256, 0, stream>>>(tbuf, P(pb + 2), P(pb + 3),
                                             nullptr, b0, nullptr, 0);
    gemm(4, b0, P(pb + 14), P(pb + 15), 64, 64, 1.0f, nullptr, zout, b3, nullptr, 0);
  };

  tri_att(2, 0);   // att_start
  tri_att(12, 1);  // att_end (transposed index maps)
  tri_mul(22, 0);  // mul_out: outgoing -> channel-major [t,c,i,k]
  tri_mul(38, 1);  // mul_in : incoming -> spatially transposed store

  // transition: 54 ln_g, 55 ln_b, 56 w1(128x64), 57 b1, 58 w2(64x128), 59 b2
  ln_kernel<<<RTOT / 16, 256, 0, stream>>>(zout, P(54), P(55), nullptr, b0,
                                           nullptr, 0);
  gemm(2, b0, P(56), P(57), 64, 128, 1.0f, (f16*)tbuf, nullptr, nullptr, nullptr, 0);
  gemm(5, (f16*)tbuf, P(58), P(59), 128, 64, 1.0f, nullptr, zout, nullptr, maskp, 0);
}
